// DNRI_Decoder_67164698575425
// MI455X (gfx1250) — matrix operand order repaired and checked by reading, not compiled, not run on hardware
//
#include <hip/hip_runtime.h>
#include <math.h>

#define B_ 32
#define N_ 512
#define E_ 32768
#define H_ 64
#define L_ 2
#define T_ 2
#define IN_ 8
#define TAU_ 0.5f

typedef __attribute__((ext_vector_type(16))) _Float16 v16h;
typedef __attribute__((ext_vector_type(8)))  float    v8f;
typedef unsigned int uint_t;

union AF { v16h v; uint_t u[8]; _Float16 h[16]; };

__device__ __forceinline__ v8f wmma_f16(v16h a, v16h b, v8f c) {
  // D = A(16x32 f16) * B(32x16 f16) + C(16x16 f32)
  return __builtin_amdgcn_wmma_f32_16x16x32_f16(false, a, false, b, (short)0, c,
                                                false, false);
}

__device__ __forceinline__ float softplusf(float x) {
  return (x > 20.0f) ? x : log1pf(expf(x));
}

// ---------------- hard gumbel-softmax (T=2), written into d_out edges region
__global__ void k_edges(const float* __restrict__ logits,
                        const float* __restrict__ gum,
                        float* __restrict__ edges) {
  int idx = blockIdx.x * 256 + threadIdx.x;      // over B*E
  float s0 = (logits[idx * 2 + 0] + gum[idx * 2 + 0]) * (1.0f / TAU_);
  float s1 = (logits[idx * 2 + 1] + gum[idx * 2 + 1]) * (1.0f / TAU_);
  float mx = fmaxf(s0, s1);
  float e0 = expf(s0 - mx), e1 = expf(s1 - mx);
  float inv = 1.0f / (e0 + e1);
  float y0 = e0 * inv, y1 = e1 * inv;
  float h0 = (s0 >= s1) ? 1.0f : 0.0f;           // argmax, first-max tie rule
  float h1 = 1.0f - h0;
  edges[idx * 2 + 0] = (h0 - y0) + y0;
  edges[idx * 2 + 1] = (h1 - y1) + y1;
}

// ---------------- 1/deg per receiver node (single block scan, E*N trivial)
__global__ void k_invdeg(const int* __restrict__ recv, float* __restrict__ invdeg) {
  int n = threadIdx.x;                            // N_ threads
  int cnt = 0;
  for (int e = 0; e < E_; ++e) cnt += (recv[e] == n) ? 1 : 0;
  invdeg[n] = 1.0f / (float)(cnt == 0 ? 1 : cnt);
}

// ---------------- f32 -> f16 convert
__global__ void k_cvt(const float* __restrict__ s, _Float16* __restrict__ d, int n) {
  int i = blockIdx.x * 256 + threadIdx.x;
  if (i < n) d[i] = (_Float16)s[i];
}

// ---------------- WMMA message-passing layer:
// per wave: 16 edges, pre=[h[recv]|h[send]] (K=128), two edge types,
// GEMM1 16 wmma + GEMM2 8 wmma per type, scatter-add (pre-scaled by invdeg).
__global__ void __launch_bounds__(256) k_msg(
    const _Float16* __restrict__ hsrc,   // [B,N,H] f16
    const _Float16* __restrict__ w1,     // [T,H,2H] f16 (this layer)
    const _Float16* __restrict__ w2,     // [T,H,H]  f16 (this layer)
    const float* __restrict__ b1,        // [T,H]
    const float* __restrict__ b2,        // [T,H]
    const float* __restrict__ edges,     // [B,E,T]
    const int* __restrict__ recv,
    const int* __restrict__ send,
    const float* __restrict__ invdeg,    // [N]
    float* __restrict__ agg)             // [B,N,H], pre-zeroed
{
  const int wv   = threadIdx.x >> 5;
  const int lane = threadIdx.x & 31;
  const int tile = blockIdx.x * 8 + wv;           // 65536 tiles total
  const int b    = tile >> 11;                    // E/16 = 2048 tiles per batch
  const int base = (tile & 2047) << 4;

  const int m  = lane & 15;                       // fragment row / column id
  const int kh = lane >> 4;                       // K-half select

  // A-matrix gather: edge m's recv/send rows
  const int e_a = base + m;
  const int rn = recv[e_a], sn = send[e_a];
  const _Float16* rrow = hsrc + ((size_t)b * N_ + rn) * H_;
  const _Float16* srow = hsrc + ((size_t)b * N_ + sn) * H_;

  AF a[4];                                        // 4 K-steps of 32
#pragma unroll
  for (int s = 0; s < 4; ++s) {
    const _Float16* row = (s < 2) ? rrow : srow;  // K<64 -> recv, K>=64 -> send
    const int koff = (s & 1) * 32;
#pragma unroll
    for (int i = 0; i < 8; ++i) {
      // contiguous K layout (matches B): dword i holds K pair kh*16 + 2i
      int kk = kh * 16 + 2 * i;
      a[s].u[i] = *(const uint_t*)(row + koff + kk);
    }
  }

  __shared__ _Float16 m1buf[8][16][64];           // per-wave C->A transpose bounce

  v8f msg[4];
#pragma unroll
  for (int c = 0; c < 4; ++c) msg[c] = (v8f){};

#pragma unroll
  for (int t = 0; t < T_; ++t) {
    // ---- GEMM1: [16x128] x [128x64]
    v8f acc[4];
#pragma unroll
    for (int c = 0; c < 4; ++c) acc[c] = (v8f){};
#pragma unroll
    for (int s = 0; s < 4; ++s) {
#pragma unroll
      for (int c = 0; c < 4; ++c) {
        const int hcol = c * 16 + m;
        const _Float16* wr = w1 + ((size_t)t * H_ + hcol) * (2 * H_) + s * 32 + kh * 16;
        v16h bf = *(const v16h*)wr;               // 16 contiguous K for column hcol
        acc[c] = wmma_f16(a[s].v, bf, acc[c]);
      }
    }
    // ---- bias + tanh, spill to LDS in [edge][h] order
#pragma unroll
    for (int c = 0; c < 4; ++c) {
      const int hcol = c * 16 + m;
      const float bias = b1[t * H_ + hcol];
#pragma unroll
      for (int r = 0; r < 8; ++r) {
        m1buf[wv][r + 8 * kh][hcol] = (_Float16)tanhf(acc[c][r] + bias);
      }
    }
    // ---- reload as A fragments (K=64 -> 2 K-steps)
    AF a2[2];
    const _Float16* mrow = &m1buf[wv][m][0];
#pragma unroll
    for (int s = 0; s < 2; ++s) {
#pragma unroll
      for (int i = 0; i < 8; ++i) {
        int kk = kh * 16 + 2 * i;
        int hh = s * 32 + kk;
        a2[s].h[2 * i + 0] = mrow[hh + 0];
        a2[s].h[2 * i + 1] = mrow[hh + 1];
      }
    }
    // ---- GEMM2: [16x64] x [64x64]
    v8f acc2[4];
#pragma unroll
    for (int c = 0; c < 4; ++c) acc2[c] = (v8f){};
#pragma unroll
    for (int s = 0; s < 2; ++s) {
#pragma unroll
      for (int c = 0; c < 4; ++c) {
        const int g = c * 16 + m;
        const _Float16* wr = w2 + ((size_t)t * H_ + g) * H_ + s * 32 + kh * 16;
        v16h bf = *(const v16h*)wr;
        acc2[c] = wmma_f16(a2[s].v, bf, acc2[c]);
      }
    }
    // ---- per-edge gumbel weight (/norm) and accumulate over types
    float wr8[8];
#pragma unroll
    for (int r = 0; r < 8; ++r) {
      int e = base + r + 8 * kh;
      wr8[r] = edges[((size_t)b * E_ + e) * T_ + t] * (1.0f / (float)T_);
    }
#pragma unroll
    for (int c = 0; c < 4; ++c) {
      const float bias = b2[t * H_ + c * 16 + m];
#pragma unroll
      for (int r = 0; r < 8; ++r) {
        msg[c][r] += tanhf(acc2[c][r] + bias) * wr8[r];
      }
    }
  }

  // ---- scatter-mean onto receiver nodes (scale by 1/deg before atomic add)
#pragma unroll
  for (int r = 0; r < 8; ++r) {
    int e = base + r + 8 * kh;
    int node = recv[e];
    float sc = invdeg[node];
    float* dst = agg + ((size_t)b * N_ + node) * H_;
#pragma unroll
    for (int c = 0; c < 4; ++c) {
      atomicAdd(dst + c * 16 + m, msg[c][r] * sc);
    }
  }
}

// ---------------- fused GRU gates + node MLP; one block per (b,n)
__global__ void __launch_bounds__(64) k_gru(
    const float* __restrict__ inputs, const float* __restrict__ hidden,
    const float* __restrict__ agg1, const float* __restrict__ agg2,
    const float* __restrict__ Whr, const float* __restrict__ Whi,
    const float* __restrict__ Whh,
    const float* __restrict__ Wir, const float* __restrict__ bir,
    const float* __restrict__ Wii, const float* __restrict__ bii,
    const float* __restrict__ Win, const float* __restrict__ binw,
    const float* __restrict__ Wm1, const float* __restrict__ bm1,
    const float* __restrict__ Wm2, const float* __restrict__ bm2,
    float* __restrict__ hidden_out, float* __restrict__ flat)
{
  const int bn = blockIdx.x;                      // b*N + n
  const int o  = threadIdx.x;
  __shared__ float sin_[IN_], sa[2 * H_], shn[H_], sm1[H_];
  const size_t rowoff = (size_t)bn * H_;
  if (o < IN_) sin_[o] = inputs[(size_t)bn * IN_ + o];
  sa[o]       = agg1[rowoff + o];
  sa[H_ + o]  = agg2[rowoff + o];
  __syncthreads();

  float xr = bir[o], xi = bii[o], xn = binw[o];
#pragma unroll
  for (int f = 0; f < IN_; ++f) {
    float v = sin_[f];
    xr += v * Wir[o * IN_ + f];
    xi += v * Wii[o * IN_ + f];
    xn += v * Win[o * IN_ + f];
  }
  float hr = 0.f, hi = 0.f, hh = 0.f;
  for (int f = 0; f < 2 * H_; ++f) {
    float v = sa[f];
    hr += v * Whr[o * (2 * H_) + f];
    hi += v * Whi[o * (2 * H_) + f];
    hh += v * Whh[o * (2 * H_) + f];
  }
  float r  = 1.f / (1.f + expf(-(xr + hr)));
  float ig = 1.f / (1.f + expf(-(xi + hi)));
  float nn = tanhf(xn + r * hh);
  float hnew = (1.f - ig) * nn + ig * hidden[rowoff + o];
  hidden_out[rowoff + o] = hnew;
  shn[o] = hnew;
  __syncthreads();

  float v1 = bm1[o];
  for (int f = 0; f < H_; ++f) v1 += shn[f] * Wm1[o * H_ + f];
  v1 = fmaxf(v1, 0.f);
  sm1[o] = v1;
  __syncthreads();

  float p = bm2[o];
  for (int f = 0; f < H_; ++f) p += sm1[f] * Wm2[o * H_ + f];
  flat[(size_t)bn * H_ + o] = p;                  // == flat[b, n*H+o]
}

// ---------------- StudentT heads: one block per (n, head); stream W row once
__global__ void __launch_bounds__(256) k_head(
    const float* __restrict__ flat,               // [B, N*H]
    const float* __restrict__ Wdf, const float* __restrict__ bdf,
    const float* __restrict__ Wloc, const float* __restrict__ bloc,
    const float* __restrict__ Wsc, const float* __restrict__ bsc,
    float* __restrict__ out)                      // df | loc | scale
{
  const int n = blockIdx.x;
  const int head = blockIdx.y;
  const float* W  = (head == 0) ? Wdf : (head == 1) ? Wloc : Wsc;
  const float* bb = (head == 0) ? bdf : (head == 1) ? bloc : bsc;
  const int tid = threadIdx.x;

  float acc[32];
#pragma unroll
  for (int b = 0; b < 32; ++b) acc[b] = 0.f;
  const float* wrow = W + (size_t)n * (N_ * H_);
  for (int j = tid; j < N_ * H_; j += 256) {
    float w = wrow[j];
#pragma unroll
    for (int b = 0; b < 32; ++b) acc[b] += w * flat[(size_t)b * (N_ * H_) + j];
  }

  __shared__ float red[256];
  for (int b = 0; b < 32; ++b) {
    red[tid] = acc[b];
    __syncthreads();
    for (int s = 128; s > 0; s >>= 1) {
      if (tid < s) red[tid] += red[tid + s];
      __syncthreads();
    }
    if (tid == 0) {
      float x = red[0] + bb[n];
      float res;
      if (head == 0)      res = 2.0f + softplusf(x);
      else if (head == 1) res = x;
      else                res = softplusf(x);
      out[head * (B_ * N_) + b * N_ + n] = res;
    }
    __syncthreads();
  }
}

extern "C" void kernel_launch(void* const* d_in, const int* in_sizes, int n_in,
                              void* d_out, int out_size, void* d_ws, size_t ws_size,
                              hipStream_t stream) {
  (void)in_sizes; (void)n_in; (void)out_size; (void)ws_size;

  const float* inputs = (const float*)d_in[0];
  const float* hidden = (const float*)d_in[1];
  const float* elog   = (const float*)d_in[2];
  const float* gum    = (const float*)d_in[3];
  const int*   send   = (const int*)d_in[4];
  const int*   recv   = (const int*)d_in[5];
  const float* W1     = (const float*)d_in[6];
  const float* b1     = (const float*)d_in[7];
  const float* W2     = (const float*)d_in[8];
  const float* b2     = (const float*)d_in[9];
  const float* Whr    = (const float*)d_in[10];
  const float* Whi    = (const float*)d_in[11];
  const float* Whh    = (const float*)d_in[12];
  const float* Wir    = (const float*)d_in[13];
  const float* bir    = (const float*)d_in[14];
  const float* Wii    = (const float*)d_in[15];
  const float* bii    = (const float*)d_in[16];
  const float* Win    = (const float*)d_in[17];
  const float* binw   = (const float*)d_in[18];
  const float* Wm1    = (const float*)d_in[19];
  const float* bm1    = (const float*)d_in[20];
  const float* Wm2    = (const float*)d_in[21];
  const float* bm2    = (const float*)d_in[22];
  const float* Wdf    = (const float*)d_in[23];
  const float* bdf    = (const float*)d_in[24];
  const float* Wloc   = (const float*)d_in[25];
  const float* bloc   = (const float*)d_in[26];
  const float* Wsc    = (const float*)d_in[27];
  const float* bsc    = (const float*)d_in[28];

  float* out        = (float*)d_out;
  float* out_df     = out;                        // df|loc|scale : 3*16384
  float* out_hidden = out + 3 * B_ * N_;          // 1048576
  float* out_edges  = out + 3 * B_ * N_ + B_ * N_ * H_;  // 2097152

  char* ws = (char*)d_ws;
  size_t off = 0;
  _Float16* hf16    = (_Float16*)(ws + off); off += (size_t)B_ * N_ * H_ * 2;
  float*    agg1    = (float*)   (ws + off); off += (size_t)B_ * N_ * H_ * 4;
  _Float16* agg1f16 = (_Float16*)(ws + off); off += (size_t)B_ * N_ * H_ * 2;
  float*    agg2    = (float*)   (ws + off); off += (size_t)B_ * N_ * H_ * 4;
  _Float16* w1f16   = (_Float16*)(ws + off); off += (size_t)L_ * T_ * H_ * 2 * H_ * 2;
  _Float16* w2f16   = (_Float16*)(ws + off); off += (size_t)L_ * T_ * H_ * H_ * 2;
  float*    invdeg  = (float*)   (ws + off); off += (size_t)N_ * 4;
  float*    flat    = (float*)   (ws + off); off += (size_t)B_ * N_ * H_ * 4;

  hipMemsetAsync(agg1, 0, (size_t)B_ * N_ * H_ * 4, stream);
  hipMemsetAsync(agg2, 0, (size_t)B_ * N_ * H_ * 4, stream);

  k_edges<<<(B_ * E_) / 256, 256, 0, stream>>>(elog, gum, out_edges);
  k_invdeg<<<1, N_, 0, stream>>>(recv, invdeg);
  k_cvt<<<(L_ * T_ * H_ * 2 * H_ + 255) / 256, 256, 0, stream>>>(W1, w1f16, L_ * T_ * H_ * 2 * H_);
  k_cvt<<<(L_ * T_ * H_ * H_ + 255) / 256, 256, 0, stream>>>(W2, w2f16, L_ * T_ * H_ * H_);
  k_cvt<<<(B_ * N_ * H_) / 256, 256, 0, stream>>>(hidden, hf16, B_ * N_ * H_);

  // layer 0: hidden -> agg1
  k_msg<<<(B_ * (E_ / 16)) / 8, 256, 0, stream>>>(hf16, w1f16, w2f16, b1, b2,
                                                  out_edges, recv, send, invdeg, agg1);
  k_cvt<<<(B_ * N_ * H_) / 256, 256, 0, stream>>>(agg1, agg1f16, B_ * N_ * H_);
  // layer 1: agg1 -> agg2
  k_msg<<<(B_ * (E_ / 16)) / 8, 256, 0, stream>>>(agg1f16,
                                                  w1f16 + (size_t)T_ * H_ * 2 * H_,
                                                  w2f16 + (size_t)T_ * H_ * H_,
                                                  b1 + T_ * H_, b2 + T_ * H_,
                                                  out_edges, recv, send, invdeg, agg2);

  k_gru<<<B_ * N_, 64, 0, stream>>>(inputs, hidden, agg1, agg2, Whr, Whi, Whh,
                                    Wir, bir, Wii, bii, Win, binw,
                                    Wm1, bm1, Wm2, bm2, out_hidden, flat);

  dim3 hg(N_, 3);
  k_head<<<hg, 256, 0, stream>>>(flat, Wdf, bdf, Wloc, bloc, Wsc, bsc, out_df);
}
